// LePEAttention_50723563765816
// MI455X (gfx1250) — compile-verified, matrix operand order checked by reading, not run on hardware
//
#include <hip/hip_runtime.h>
#include <hip/hip_bf16.h>

// ---------------- problem constants ----------------
constexpr int kB = 2, kT = 4, kH = 56, kW = 56;
constexpr int kTSP = 4, kHSP = 56, kWSP = 7;
constexpr int kC = 64, kHEADS = 2, kHD = 32;
constexpr int kS = kTSP * kHSP * kWSP;          // 1568 tokens / window
constexpr int kL = kT * kH * kW;                // 12544 tokens / batch
constexpr long long kPlane = (long long)kB * kL * kC;  // q->k->v plane stride
constexpr float kScale = 0.17677669529663687f;  // 32^-0.5
constexpr int kNW = kB * (kW / kWSP);           // 16 windows total

// tiling: each wave owns 32 q rows (two 16-row WMMA tiles)
constexpr int BM = 224;                 // q rows per block (1568 = 7*224)
constexpr int BN = 32;                  // kv tokens per iteration (1568 = 49*32)
constexpr int NKV = kS / BN;            // 49
constexpr int WAVES = 7;
constexpr int NT = WAVES * 32;          // 224 threads

// ---------------- vector types ----------------
typedef __attribute__((ext_vector_type(4)))  unsigned short v4us;
typedef __attribute__((ext_vector_type(8)))  unsigned short v8us;
typedef __attribute__((ext_vector_type(16))) unsigned short v16us;
typedef __attribute__((ext_vector_type(16))) __bf16        v16bf;
typedef __attribute__((ext_vector_type(8)))  float         v8f;

// ---------------- helpers ----------------
__device__ __forceinline__ unsigned short f2bf(float f) {
  unsigned int u = __float_as_uint(f);
  u += 0x7FFFu + ((u >> 16) & 1u);       // round-to-nearest-even
  return (unsigned short)(u >> 16);
}

// window token s -> flat l inside (T,H,W) image for W-split block wb
__device__ __forceinline__ int tok2l(int s, int wb) {
  int t  = s / (kHSP * kWSP);
  int r  = s - t * (kHSP * kWSP);
  int h  = r / kWSP;
  int ww = r - h * kWSP;
  return t * (kH * kW) + h * kW + wb * kWSP + ww;
}

// gather one 16-bf16 WMMA fragment lane-slice from a 32-element LDS row.
// lanes 0-15 use b0=0 (K 0-7,16-23); lanes 16-31 use b0=8 (K 8-15,24-31).
__device__ __forceinline__ v16bf frag16(const unsigned short* row, int b0) {
  v8us lo = *reinterpret_cast<const v8us*>(row + b0);
  v8us hi = *reinterpret_cast<const v8us*>(row + 16 + b0);
  v16us w = __builtin_shufflevector(lo, hi, 0,1,2,3,4,5,6,7,8,9,10,11,12,13,14,15);
  return __builtin_bit_cast(v16bf, w);
}

// single-instruction max (avoids llvm.maxnum canonicalize padding)
__device__ __forceinline__ float vmaxf(float a, float b) {
  float r;
  asm("v_max_num_f32 %0, %1, %2" : "=v"(r) : "v"(a), "v"(b));
  return r;
}
// fused DPP row_ror butterfly max: one VOP2+DPP instruction per step.
// Reduces across each 16-lane row (the WMMA C-layout N stripe).
__device__ __forceinline__ float rmax16(float v) {
  float t;
  asm("v_max_num_f32_dpp %0, %1, %1 row_ror:1" : "=v"(t) : "v"(v));
  asm("v_max_num_f32_dpp %0, %1, %1 row_ror:2" : "=v"(v) : "v"(t));
  asm("v_max_num_f32_dpp %0, %1, %1 row_ror:4" : "=v"(t) : "v"(v));
  asm("v_max_num_f32_dpp %0, %1, %1 row_ror:8" : "=v"(v) : "v"(t));
  return v;
}

// ---------------- LePE: depthwise 3x3x3 conv on per-window V ----------------
__global__ void cswin_lepe_kernel(const float* __restrict__ qkv,
                                  const float* __restrict__ w,
                                  const float* __restrict__ bias,
                                  float* __restrict__ lepe) {
  int idx  = blockIdx.x * blockDim.x + threadIdx.x;   // (wi*S + s)*C + c
  int c    = idx & (kC - 1);
  int rest = idx >> 6;
  int s    = rest % kS;
  int wi   = rest / kS;
  int b  = wi >> 3, wb = wi & 7;
  int t  = s / (kHSP * kWSP);
  int r  = s - t * (kHSP * kWSP);
  int h  = r / kWSP;
  int ww = r - h * kWSP;

  const float* vplane = qkv + 2 * kPlane + (long long)b * kL * kC;
  float acc = bias[c];
  #pragma unroll
  for (int dt = 0; dt < 3; ++dt) {
    int tt = t + dt - 1;
    if ((unsigned)tt >= (unsigned)kTSP) continue;
    #pragma unroll
    for (int dh = 0; dh < 3; ++dh) {
      int hh = h + dh - 1;
      if ((unsigned)hh >= (unsigned)kHSP) continue;
      #pragma unroll
      for (int dw = 0; dw < 3; ++dw) {
        int www = ww + dw - 1;
        if ((unsigned)www >= (unsigned)kWSP) continue;
        int l = tt * (kH * kW) + hh * kW + wb * kWSP + www;
        acc += w[c * 27 + dt * 9 + dh * 3 + dw] * vplane[(long long)l * kC + c];
      }
    }
  }
  lepe[idx] = acc;
}

// ---------------- fused flash attention per (window, head) ----------------
__global__ __launch_bounds__(NT, 1)
void cswin_flash_attn(const float* __restrict__ qkv,
                      const float* __restrict__ lepe,
                      float* __restrict__ out) {
  __shared__ __align__(16) unsigned short Qs[BM * 32];          // bf16, pre-scaled
  __shared__ __align__(16) unsigned short Ks[2][BN * 32];       // token-major, dbl-buf
  __shared__ __align__(16) unsigned short VTs[2][32 * BN];      // dim-major, dbl-buf
  __shared__ __align__(16) unsigned short Ps[WAVES * 32 * 32];  // per-wave P scratch

  const int tid  = threadIdx.x;
  const int wave = tid >> 5;
  const int lane = tid & 31;
  const int wh   = blockIdx.x;        // 0..31
  const int wi   = wh >> 1;
  const int head = wh & 1;
  const int b    = wi >> 3;
  const int wb   = wi & 7;
  const int qrow0 = blockIdx.y * BM;

  const long long batchBase = (long long)b * kL * kC;
  const int chBase = head * kHD;

  // ---- stage Q tile (scaled, bf16): 224 rows x 32 dims = 1792 float4 groups
  #pragma unroll
  for (int i = 0; i < 8; ++i) {
    int g    = tid + i * NT;               // 0..1791
    int row  = g >> 3;
    int dim4 = (g & 7) * 4;
    int l    = tok2l(qrow0 + row, wb);
    const float4 f = *reinterpret_cast<const float4*>(
        qkv + batchBase + (long long)l * kC + chBase + dim4);
    v4us pk = { f2bf(f.x * kScale), f2bf(f.y * kScale),
                f2bf(f.z * kScale), f2bf(f.w * kScale) };
    *reinterpret_cast<v4us*>(&Qs[row * 32 + dim4]) = pk;   // ds_store_b64
  }

  // ---- K/V staging helpers (512 float4 groups -> 3 strided slots/thread)
  float4 kvreg[3];
  auto kv_load = [&](int kb) {
    #pragma unroll
    for (int i = 0; i < 3; ++i) {
      int g = tid + i * NT;
      if (g < 512) {
        const int isV  = g >> 8;
        const int gg   = g & 255;
        const int tok  = gg >> 3;
        const int dim4 = (gg & 7) * 4;
        const int l    = tok2l(kb * BN + tok, wb);
        kvreg[i] = *reinterpret_cast<const float4*>(
            qkv + (1 + isV) * kPlane + batchBase +
            (long long)l * kC + chBase + dim4);
      }
    }
  };
  auto kv_store = [&](int buf) {
    #pragma unroll
    for (int i = 0; i < 3; ++i) {
      int g = tid + i * NT;
      if (g < 512) {
        const int isV  = g >> 8;
        const int gg   = g & 255;
        const int tok  = gg >> 3;
        const int dim4 = (gg & 7) * 4;
        const float4 f = kvreg[i];
        if (!isV) {
          v4us pk = { f2bf(f.x), f2bf(f.y), f2bf(f.z), f2bf(f.w) };
          *reinterpret_cast<v4us*>(&Ks[buf][tok * 32 + dim4]) = pk;
        } else {
          VTs[buf][(dim4 + 0) * BN + tok] = f2bf(f.x);
          VTs[buf][(dim4 + 1) * BN + tok] = f2bf(f.y);
          VTs[buf][(dim4 + 2) * BN + tok] = f2bf(f.z);
          VTs[buf][(dim4 + 3) * BN + tok] = f2bf(f.w);
        }
      }
    }
  };

  // prologue: fill buffer 0
  kv_load(0);
  kv_store(0);
  __syncthreads();

  const int b0 = (lane & 16) ? 8 : 0;
  const int nl = lane & 15;
  v16bf aQ[2];
  #pragma unroll
  for (int mi = 0; mi < 2; ++mi)
    aQ[mi] = frag16(&Qs[(wave * 32 + mi * 16 + nl) * 32], b0);

  // all-ones bf16 B fragment: P x ones = per-row sums via WMMA
  const v16bf bOnes = __builtin_bit_cast(v16bf, (v16us)((unsigned short)0x3F80));

  float mmax[2][8], lsum[2][8];
  v8f O[2][2];
  #pragma unroll
  for (int mi = 0; mi < 2; ++mi) {
    #pragma unroll
    for (int j = 0; j < 8; ++j) { mmax[mi][j] = -3.0e38f; lsum[mi][j] = 0.0f; }
    O[mi][0] = v8f{0,0,0,0,0,0,0,0};
    O[mi][1] = v8f{0,0,0,0,0,0,0,0};
  }

  unsigned short* pbase = &Ps[wave * 1024];

  for (int kb = 0; kb < NKV; ++kb) {
    const int cur = kb & 1;
    // issue next block's global loads first; compute hides their latency
    if (kb + 1 < NKV) kv_load(kb + 1);
    if (kb + 2 < NKV) {   // gfx1250 prefetch two blocks ahead
      const int l2 = tok2l((kb + 2) * BN + ((tid >> 3) & 31), wb);
      __builtin_prefetch(qkv + kPlane + batchBase + (long long)l2 * kC + chBase, 0, 3);
    }

    const unsigned short* kbuf = &Ks[cur][0];
    const unsigned short* vbuf = &VTs[cur][0];

    // ---- scores: S = Q * K^T   (2 M-tiles x 2 N-tiles, K-dim = hd = 32)
    const v16bf bK0 = frag16(&kbuf[nl * 32], b0);
    const v16bf bK1 = frag16(&kbuf[(16 + nl) * 32], b0);
    const v8f z = {0,0,0,0,0,0,0,0};
    v8f sc[2][2];
    sc[0][0] = __builtin_amdgcn_wmma_f32_16x16x32_bf16(false, aQ[0], false, bK0,
                                                       (short)0, z, false, false);
    sc[0][1] = __builtin_amdgcn_wmma_f32_16x16x32_bf16(false, aQ[0], false, bK1,
                                                       (short)0, z, false, false);
    sc[1][0] = __builtin_amdgcn_wmma_f32_16x16x32_bf16(false, aQ[1], false, bK0,
                                                       (short)0, z, false, false);
    sc[1][1] = __builtin_amdgcn_wmma_f32_16x16x32_bf16(false, aQ[1], false, bK1,
                                                       (short)0, z, false, false);

    // ---- online softmax: fused DPP max butterflies, exp, rescale, stage P
    float scl[2][8];
    #pragma unroll
    for (int mi = 0; mi < 2; ++mi) {
      #pragma unroll
      for (int j = 0; j < 8; ++j) {
        const float tmax = rmax16(vmaxf(sc[mi][0][j], sc[mi][1][j]));
        const float nm   = vmaxf(mmax[mi][j], tmax);
        scl[mi][j] = __expf(mmax[mi][j] - nm);
        mmax[mi][j] = nm;
        const float p0 = __expf(sc[mi][0][j] - nm);
        const float p1 = __expf(sc[mi][1][j] - nm);
        O[mi][0][j] *= scl[mi][j];
        O[mi][1][j] *= scl[mi][j];
        const int m = j + ((lane & 16) ? 8 : 0);
        pbase[(mi * 16 + m) * 32 + nl]      = f2bf(p0);
        pbase[(mi * 16 + m) * 32 + 16 + nl] = f2bf(p1);
      }
    }

    // ---- O += P * V, and row sums via P x ones (per-wave LDS: in-order DS)
    const v16bf aP0 = frag16(&pbase[nl * 32], b0);
    const v16bf aP1 = frag16(&pbase[(16 + nl) * 32], b0);
    const v16bf bV0 = frag16(&vbuf[nl * BN], b0);
    const v16bf bV1 = frag16(&vbuf[(16 + nl) * BN], b0);
    O[0][0] = __builtin_amdgcn_wmma_f32_16x16x32_bf16(false, aP0, false, bV0,
                                                      (short)0, O[0][0], false, false);
    O[0][1] = __builtin_amdgcn_wmma_f32_16x16x32_bf16(false, aP0, false, bV1,
                                                      (short)0, O[0][1], false, false);
    O[1][0] = __builtin_amdgcn_wmma_f32_16x16x32_bf16(false, aP1, false, bV0,
                                                      (short)0, O[1][0], false, false);
    O[1][1] = __builtin_amdgcn_wmma_f32_16x16x32_bf16(false, aP1, false, bV1,
                                                      (short)0, O[1][1], false, false);
    v8f rs0 = __builtin_amdgcn_wmma_f32_16x16x32_bf16(false, aP0, false, bOnes,
                                                      (short)0, z, false, false);
    v8f rs1 = __builtin_amdgcn_wmma_f32_16x16x32_bf16(false, aP1, false, bOnes,
                                                      (short)0, z, false, false);
    #pragma unroll
    for (int j = 0; j < 8; ++j) {
      lsum[0][j] = lsum[0][j] * scl[0][j] + rs0[j];
      lsum[1][j] = lsum[1][j] * scl[1][j] + rs1[j];
    }

    // flush next block's staged regs into the other LDS buffer; one barrier
    if (kb + 1 < NKV) kv_store(cur ^ 1);
    __syncthreads();
  }

  // ---- epilogue: normalize, add LePE, scatter back through windows2img
  #pragma unroll
  for (int mi = 0; mi < 2; ++mi) {
    #pragma unroll
    for (int j = 0; j < 8; ++j) {
      const int m    = j + ((lane & 16) ? 8 : 0);
      const int srow = qrow0 + wave * 32 + mi * 16 + m;
      const int l    = tok2l(srow, wb);
      const float inv = 1.0f / lsum[mi][j];
      const int c0 = chBase + nl;
      const int c1 = c0 + 16;
      const long long ob = batchBase + (long long)l * kC;
      const long long lb = ((long long)wi * kS + srow) * kC;
      out[ob + c0] = O[mi][0][j] * inv + lepe[lb + c0];
      out[ob + c1] = O[mi][1][j] * inv + lepe[lb + c1];
    }
  }
}

// ---------------- launch ----------------
extern "C" void kernel_launch(void* const* d_in, const int* in_sizes, int n_in,
                              void* d_out, int out_size, void* d_ws, size_t ws_size,
                              hipStream_t stream) {
  (void)in_sizes; (void)n_in; (void)out_size; (void)ws_size;
  const float* qkv    = (const float*)d_in[0];
  const float* lepe_w = (const float*)d_in[1];
  const float* lepe_b = (const float*)d_in[2];
  float* out  = (float*)d_out;
  float* lepe = (float*)d_ws;   // 16*1568*64 floats = 6.4 MB scratch

  const int lepeElems = kNW * kS * kC;
  cswin_lepe_kernel<<<lepeElems / 256, 256, 0, stream>>>(qkv, lepe_w, lepe_b, lepe);

  dim3 grid(kNW * kHEADS, kS / BM);     // 32 x 7
  cswin_flash_attn<<<grid, NT, 0, stream>>>(qkv, lepe, out);
}
